// DeformableGConv_58634893525710
// MI455X (gfx1250) — matrix operand order repaired and checked by reading, not compiled, not run on hardware
//
#include <hip/hip_runtime.h>
#include <hip/hip_bf16.h>
#include <math.h>

#define N_NODES 50000
#define N_EDGES 800000
#define RAW     128
#define HIDDEN  256
#define OUTD    64
#define KK      4
#define PHID    16
#define EPSF    1e-8f

typedef __attribute__((ext_vector_type(16))) __bf16 v16bf;
typedef __attribute__((ext_vector_type(8)))  float  v8f;

union BF16Frag { v16bf v; __bf16 h[16]; };

// native f32 -> bf16 converts (lowered to v_cvt_pk_bf16_f32-class ops)
__device__ __forceinline__ void cvt_f4(const float4 f, __bf16* d) {
  d[0] = (__bf16)f.x; d[1] = (__bf16)f.y; d[2] = (__bf16)f.z; d[3] = (__bf16)f.w;
}

__device__ __forceinline__ void atomicMaxF(float* addr, float val) {
  unsigned* ua = (unsigned*)addr;
  unsigned cur = __float_as_uint(*addr);
  while (__uint_as_float(cur) < val) {
    unsigned prev = atomicCAS(ua, cur, __float_as_uint(val));
    if (prev == cur) break;
    cur = prev;
  }
}

// ---------------------------------------------------------------------------
// WMMA GEMM: C[M x Nn] = act(A[M x Kd] * B + bias)
//   B_KXN=true : B is stored [Kd x Nn] row-major (plain  A@B)
//   B_KXN=false: B is stored [Nn x Kd] row-major (i.e. A@B.T, B=weights)
// One wave per 16x16 output tile; fp32 loaded as float4, converted to bf16 in
// registers; V_WMMA_F32_16X16X32_BF16 accumulates in f32.
// ---------------------------------------------------------------------------
template<bool B_KXN, bool RELU>
__global__ __launch_bounds__(256)
void wmma_gemm_kernel(const float* __restrict__ A,
                      const float* __restrict__ B,
                      const float* __restrict__ bias,
                      float* __restrict__ C,
                      int M, int Nn, int Kd) {
  const int wave = threadIdx.x >> 5;
  const int lane = threadIdx.x & 31;
  const int tilesN = Nn >> 4;
  const int tile = blockIdx.x * (blockDim.x >> 5) + wave;
  const int totalTiles = (M >> 4) * tilesN;
  if (tile >= totalTiles) return;              // wave-uniform exit (EXEC stays all-1)
  const int mt = tile / tilesN;
  const int nt = tile - mt * tilesN;

  const int half = lane >> 4;                  // 0: lanes 0-15, 1: lanes 16-31
  const int l16  = lane & 15;
  const int rowA = (mt << 4) + l16;            // A row held by this lane
  const int colB = (nt << 4) + l16;            // B column held by this lane

  v8f acc = {};
  for (int k0 = 0; k0 < Kd; k0 += 32) {
    BF16Frag a, b;
    // A fragment (16x32 bf16): lane<16 -> K {0..15}; lane>=16 -> K {16..31}
    const float* Arow = A + (size_t)rowA * Kd;
    const int kA0 = k0 + half * 16;            // 32B-aligned
    const int kA1 = kA0 + 8;
    cvt_f4(*(const float4*)(Arow + kA0),     &a.h[0]);
    cvt_f4(*(const float4*)(Arow + kA0 + 4), &a.h[4]);
    cvt_f4(*(const float4*)(Arow + kA1),     &a.h[8]);
    cvt_f4(*(const float4*)(Arow + kA1 + 4), &a.h[12]);

    // B fragment (32x16 bf16): lanes 0-15 hold K=k0..k0+15, lanes 16-31 K=k0+16..k0+31
    const int kB = k0 + half * 16;
    if (B_KXN) {
      // column of row-major [Kd x Nn] weight: strided scalar loads (L2-resident)
#pragma unroll
      for (int i = 0; i < 16; ++i)
        b.h[i] = (__bf16)B[(size_t)(kB + i) * Nn + colB];
    } else {
      // row of row-major [Nn x Kd] weight: 16 contiguous floats, 64B-aligned
      const float* Brow = B + (size_t)colB * Kd + kB;
      cvt_f4(*(const float4*)(Brow),      &b.h[0]);
      cvt_f4(*(const float4*)(Brow + 4),  &b.h[4]);
      cvt_f4(*(const float4*)(Brow + 8),  &b.h[8]);
      cvt_f4(*(const float4*)(Brow + 12), &b.h[12]);
    }
    acc = __builtin_amdgcn_wmma_f32_16x16x32_bf16(
        false, a.v, false, b.v, (short)0, acc, false, false);
  }

  // C/D layout: element j -> row mt*16 + half*8 + j, col nt*16 + l16
  const float bv = bias ? bias[colB] : 0.0f;
  const int rowBase = (mt << 4) + half * 8;
#pragma unroll
  for (int j = 0; j < 8; ++j) {
    float v = acc[j] + bv;
    if (RELU) v = fmaxf(v, 0.0f);
    C[(size_t)(rowBase + j) * Nn + colB] = v;
  }
}

// ---------------------------------------------------------------------------
// phi_k = e_l @ W_phi   ([N x 128] @ [128 x 15]) — tiny; W_phi staged in LDS.
// ---------------------------------------------------------------------------
__global__ void phi_kernel(const float* __restrict__ e_l,
                           const float* __restrict__ W_phi,
                           float* __restrict__ phi) {
  __shared__ float w[RAW * (PHID - 1)];
  for (int i = threadIdx.x; i < RAW * (PHID - 1); i += blockDim.x)
    w[i] = W_phi[i];
  __syncthreads();
  int n = blockIdx.x * blockDim.x + threadIdx.x;
  if (n >= N_NODES) return;
  float acc[PHID - 1];
#pragma unroll
  for (int p = 0; p < PHID - 1; ++p) acc[p] = 0.0f;
  const float* row = e_l + (size_t)n * RAW;
  for (int k = 0; k < RAW; ++k) {
    float v = row[k];
#pragma unroll
    for (int p = 0; p < PHID - 1; ++p)
      acc[p] += v * w[k * (PHID - 1) + p];
  }
#pragma unroll
  for (int p = 0; p < PHID - 1; ++p) phi[(size_t)n * (PHID - 1) + p] = acc[p];
}

// ---------------------------------------------------------------------------
// tilde_norm + l_sep (tiny, one block of 64 threads)
// ---------------------------------------------------------------------------
__global__ void tilde_kernel(const float* __restrict__ tilde_phi,
                             float* __restrict__ tn,
                             float* __restrict__ lsep_out) {
  __shared__ float t[KK * PHID];
  __shared__ float nrm[KK];
  int tid = threadIdx.x;
  if (tid < KK * PHID) t[tid] = tilde_phi[tid];
  __syncthreads();
  if (tid < KK) {
    float s = 0.0f;
    for (int p = 0; p < PHID; ++p) s += t[tid * PHID + p] * t[tid * PHID + p];
    nrm[tid] = sqrtf(s);
  }
  __syncthreads();
  if (tid < KK * PHID) tn[tid] = t[tid] / nrm[tid / PHID];
  if (tid == 0) {
    float lsep = 0.0f;
    for (int k = 0; k < KK; ++k)
      for (int j = 0; j < KK; ++j) {
        float d2 = 0.0f;
        for (int p = 0; p < PHID; ++p) {
          float d = t[k * PHID + p] / nrm[k] - t[j * PHID + p] / nrm[j];
          d2 += d * d;
        }
        lsep += d2;
      }
    lsep_out[0] = lsep / (float)KK;
  }
}

__global__ void init_kernel(float* __restrict__ m, float* __restrict__ s,
                            float* __restrict__ outacc, float* __restrict__ lf) {
  int i = blockIdx.x * blockDim.x + threadIdx.x;
  if (i < N_NODES * KK) { m[i] = -INFINITY; s[i] = 0.0f; }
  if (i < N_NODES * OUTD) outacc[i] = 0.0f;
  if (i == 0) lf[0] = 0.0f;
}

// ---------------------------------------------------------------------------
// Edge phase
// ---------------------------------------------------------------------------
__global__ void edge_logit_kernel(const float* __restrict__ phi,
                                  const float* __restrict__ delta,
                                  const float* __restrict__ tn,
                                  const int* __restrict__ src,
                                  const int* __restrict__ dst,
                                  float* __restrict__ attn,
                                  float* __restrict__ m) {
  int e = blockIdx.x * blockDim.x + threadIdx.x;
  if (e >= N_EDGES) return;
  int si = src[e], di = dst[e];
  float d[PHID];
  bool allz = true;
  float n2 = 0.0f;
#pragma unroll
  for (int p = 0; p < PHID - 1; ++p) {
    float v = phi[(size_t)si * (PHID - 1) + p] - phi[(size_t)di * (PHID - 1) + p];
    d[p] = v;
    allz = allz && (v == 0.0f);
    n2 += v * v;
  }
  d[PHID - 1] = allz ? 1.0f : 0.0f;
  n2 += d[PHID - 1] * d[PHID - 1];
  float inv = 1.0f / fmaxf(sqrtf(n2), EPSF);
#pragma unroll
  for (int k = 0; k < KK; ++k) {
    float a = 0.0f;
#pragma unroll
    for (int p = 0; p < PHID; ++p)
      a += d[p] * inv * (tn[k * PHID + p] + delta[(size_t)di * (KK * PHID) + k * PHID + p]);
    attn[(size_t)e * KK + k] = a;
    atomicMaxF(&m[di * KK + k], a);
  }
}

__global__ void edge_exp_kernel(const int* __restrict__ dst,
                                float* __restrict__ attn,
                                const float* __restrict__ m,
                                float* __restrict__ s) {
  int i = blockIdx.x * blockDim.x + threadIdx.x;
  if (i >= N_EDGES * KK) return;
  int e = i >> 2, k = i & 3;
  int di = dst[e];
  float ex = __expf(attn[i] - m[di * KK + k]);
  attn[i] = ex;
  atomicAdd(&s[di * KK + k], ex);
}

__global__ void edge_norm_kernel(const int* __restrict__ dst,
                                 float* __restrict__ attn,
                                 const float* __restrict__ s) {
  int i = blockIdx.x * blockDim.x + threadIdx.x;
  if (i >= N_EDGES * KK) return;
  int e = i >> 2, k = i & 3;
  attn[i] = attn[i] / s[dst[e] * KK + k];
}

// One thread per (edge, out-channel); fold the k-sum, one atomicAdd per thread.
__global__ void message_kernel(const float* __restrict__ h,
                               const float* __restrict__ attn,
                               const int* __restrict__ src,
                               const int* __restrict__ dst,
                               float* __restrict__ outacc) {
  long long t = (long long)blockIdx.x * blockDim.x + threadIdx.x;
  if (t >= (long long)N_EDGES * OUTD) return;
  int e = (int)(t >> 6);
  int o = (int)(t & 63);
  int si = src[e], di = dst[e];
  float acc = 0.0f;
#pragma unroll
  for (int k = 0; k < KK; ++k)
    acc += h[(size_t)si * (KK * OUTD) + k * OUTD + o] * attn[(size_t)e * KK + k];
  atomicAdd(&outacc[(size_t)di * OUTD + o], acc);
}

__global__ void finalize_kernel(float* __restrict__ out,
                                const float* __restrict__ bias) {
  int n = blockIdx.x * blockDim.x + threadIdx.x;
  if (n >= N_NODES) return;
  float v[OUTD];
  float n2 = 0.0f;
#pragma unroll
  for (int o = 0; o < OUTD; ++o) {
    float x = out[(size_t)n * OUTD + o] + bias[o];
    v[o] = x;
    n2 += x * x;
  }
  float inv = 1.0f / fmaxf(sqrtf(n2), EPSF);
#pragma unroll
  for (int o = 0; o < OUTD; ++o) out[(size_t)n * OUTD + o] = v[o] * inv;
}

__global__ void lfocus_kernel(const float* __restrict__ delta,
                              float* __restrict__ acc) {
  __shared__ float red[256];
  long long total = (long long)N_NODES * KK * PHID;
  long long i = (long long)blockIdx.x * blockDim.x + threadIdx.x;
  float local = 0.0f;
  for (long long j = i; j < total; j += (long long)gridDim.x * blockDim.x) {
    float v = delta[j];
    local += v * v;
  }
  red[threadIdx.x] = local;
  __syncthreads();
  for (int st = 128; st > 0; st >>= 1) {
    if (threadIdx.x < st) red[threadIdx.x] += red[threadIdx.x + st];
    __syncthreads();
  }
  if (threadIdx.x == 0) atomicAdd(acc, red[0]);
}

__global__ void lfocus_write_kernel(const float* __restrict__ acc,
                                    float* __restrict__ dst) {
  dst[0] = acc[0] / (float)((long long)N_NODES * KK);
}

// ---------------------------------------------------------------------------
extern "C" void kernel_launch(void* const* d_in, const int* in_sizes, int n_in,
                              void* d_out, int out_size, void* d_ws, size_t ws_size,
                              hipStream_t stream) {
  const float* h_l   = (const float*)d_in[0];
  const float* e_l   = (const float*)d_in[1];
  const int*   src   = (const int*)  d_in[2];
  const int*   dst   = (const int*)  d_in[3];
  const float* W_phi = (const float*)d_in[4];
  const float* W1    = (const float*)d_in[5];
  const float* b1    = (const float*)d_in[6];
  const float* W2    = (const float*)d_in[7];
  const float* b2    = (const float*)d_in[8];
  const float* tilde = (const float*)d_in[9];
  const float* W     = (const float*)d_in[10];
  const float* bias  = (const float*)d_in[11];

  float* out = (float*)d_out;               // [N*OUT] ++ [l_sep] ++ [l_focus]
  float* p   = (float*)d_ws;
  float* h     = p; p += (size_t)N_NODES * HIDDEN;        // h = h_l @ W
  float* hid   = p; p += (size_t)N_NODES * HIDDEN;        // relu(e_l @ W1^T + b1)
  float* delta = p; p += (size_t)N_NODES * KK * PHID;     // hid @ W2^T + b2
  float* phi   = p; p += (size_t)N_NODES * (PHID - 1);    // e_l @ W_phi
  float* attn  = p; p += (size_t)N_EDGES * KK;            // logits -> exp -> att
  float* m     = p; p += (size_t)N_NODES * KK;            // segment max
  float* s     = p; p += (size_t)N_NODES * KK;            // segment sum
  float* tn    = p; p += 64;                              // normalized anchors
  float* lfa   = p; p += 1;                               // l_focus accumulator

  const int B256 = 256;

  // init accumulators (m=-inf, s=0, out=0, lfocus=0)
  init_kernel<<<(N_NODES * OUTD + B256 - 1) / B256, B256, 0, stream>>>(m, s, out, lfa);

  // anchors + l_sep scalar
  tilde_kernel<<<1, 64, 0, stream>>>(tilde, tn, out + (size_t)N_NODES * OUTD);

  // dense GEMMs on WMMA (one wave per 16x16 tile; 8 waves per block)
  {
    int tiles = (N_NODES / 16) * (HIDDEN / 16);           // 50000
    wmma_gemm_kernel<true, false><<<(tiles + 7) / 8, B256, 0, stream>>>(
        h_l, W, nullptr, h, N_NODES, HIDDEN, HIDDEN);     // h = h_l @ W
  }
  {
    int tiles = (N_NODES / 16) * (HIDDEN / 16);           // 50000
    wmma_gemm_kernel<false, true><<<(tiles + 7) / 8, B256, 0, stream>>>(
        e_l, W1, b1, hid, N_NODES, HIDDEN, RAW);          // hid = relu(e_l @ W1^T + b1)
  }
  {
    int tiles = (N_NODES / 16) * ((KK * PHID) / 16);      // 12500
    wmma_gemm_kernel<false, false><<<(tiles + 7) / 8, B256, 0, stream>>>(
        hid, W2, b2, delta, N_NODES, KK * PHID, HIDDEN);  // delta = hid @ W2^T + b2
  }

  // pseudo-coordinates
  phi_kernel<<<(N_NODES + B256 - 1) / B256, B256, 0, stream>>>(e_l, W_phi, phi);

  // l_focus = mean(sum(delta^2, -1))
  lfocus_kernel<<<512, B256, 0, stream>>>(delta, lfa);
  lfocus_write_kernel<<<1, 1, 0, stream>>>(lfa, out + (size_t)N_NODES * OUTD + 1);

  // edge attention
  edge_logit_kernel<<<(N_EDGES + B256 - 1) / B256, B256, 0, stream>>>(
      phi, delta, tn, src, dst, attn, m);
  edge_exp_kernel<<<(N_EDGES * KK + B256 - 1) / B256, B256, 0, stream>>>(dst, attn, m, s);
  edge_norm_kernel<<<(N_EDGES * KK + B256 - 1) / B256, B256, 0, stream>>>(dst, attn, s);

  // message passing (k-sum folded, one atomic per (dst, o))
  {
    long long threads = (long long)N_EDGES * OUTD;
    int blocks = (int)((threads + B256 - 1) / B256);
    message_kernel<<<blocks, B256, 0, stream>>>(h, attn, src, dst, out);
  }

  // bias + row L2-normalize
  finalize_kernel<<<(N_NODES + B256 - 1) / B256, B256, 0, stream>>>(out, bias);
}
